// MCRE_54022098649175
// MI455X (gfx1250) — matrix operand order repaired and checked by reading, not compiled, not run on hardware
//
#include <hip/hip_runtime.h>
#include <hip/hip_bf16.h>
#include <math.h>

// ---------------------------------------------------------------------------
// CDNA5 (gfx1250): WMMA bf16 GEMMs + TDM (tensor_load_to_lds) B-tile staging,
// LDS-resident Whh LSTM scan, bf16 activation hand-off between GEMMs.
// ---------------------------------------------------------------------------

typedef __attribute__((ext_vector_type(16))) __bf16 v16bf;
typedef __attribute__((ext_vector_type(8)))  float  v8f;
typedef __attribute__((ext_vector_type(4)))  unsigned v4u;
typedef __attribute__((ext_vector_type(8)))  int      v8i;
typedef __attribute__((ext_vector_type(4)))  int      v4i;

union Frag16 { v16bf v; unsigned u[8]; unsigned short s[16]; };

__device__ __forceinline__ unsigned short f2bf(float f) {
  unsigned x = __float_as_uint(f);
  x += 0x7FFFu + ((x >> 16) & 1u);
  return (unsigned short)(x >> 16);
}
__device__ __forceinline__ float sigm(float x) { return 1.0f / (1.0f + __expf(-x)); }

static constexpr int Bb = 8, Lb = 256, Hd = 768;
static constexpr int TBM = 64, TBN = 128, TBK = 32;
static constexpr int AP = 40, BP = 136;   // padded LDS strides

// --- TDM: 2D tile (32 rows x 128 bf16 cols) global -> LDS, 16B pad per row ---
__device__ __forceinline__ void tdm_load_b_tile(unsigned ldsoff,
                                                const unsigned short* gbase,
                                                long startElem, long rowStrideElem,
                                                unsigned tW, unsigned tH) {
  unsigned long long ga = (unsigned long long)(uintptr_t)(gbase + startElem);
  v4u g0;
  g0[0] = 1u;                                   // count=1, user descriptor
  g0[1] = ldsoff;                               // LDS byte address
  g0[2] = (unsigned)(ga & 0xFFFFFFFFull);
  g0[3] = (unsigned)((ga >> 32) & 0x01FFFFFFull) | (2u << 30);  // type=2 (image)
  v8i g1;
  // data_size=2B(1), pad_enable, pad_interval=64 DW(5), pad_amount=4 DW(3)
  g1[0] = (int)((1u << 16) | (1u << 20) | (5u << 22) | (3u << 25));
  g1[1] = (int)((tW & 0xFFFFu) << 16);                       // tensor_dim0[15:0]
  g1[2] = (int)((tW >> 16) | ((tH & 0xFFFFu) << 16));        // dim0 hi | dim1 lo
  g1[3] = (int)((tH >> 16) | (128u << 16));                  // dim1 hi | tile_dim0=128
  g1[4] = (int)32;                                           // tile_dim1=32
  unsigned long long s0 = (unsigned long long)rowStrideElem;
  g1[5] = (int)(s0 & 0xFFFFFFFFull);
  g1[6] = (int)((s0 >> 32) & 0xFFFFull);
  g1[7] = 0;
  v4i z4 = {0, 0, 0, 0};
  v8i z8 = {0, 0, 0, 0, 0, 0, 0, 0};
  __builtin_amdgcn_tensor_load_to_lds(g0, g1, z4, z4, z8, 0);
}

// ======================= templated WMMA GEMM ===============================
// AMODE: 0 row=m   1 x-permute (b=m%B,l=m/B -> b*L+l)   2 row=m*B+z
// BMODE: 0 W^T from transposed bf16 [K,N] (TDM)  1 Kh16 (row=n*B+z, coop)
//        2 V row-major bf16 (row=k*B+z, TDM)
// CMODE: 0 row=m   1 row=m*B+z
template <int AMODE, int BMODE, bool A16, bool B16, bool BTDM>
__global__ __launch_bounds__(256) void k_gemm(
    const void* __restrict__ Ap, int lda, int aofs, long aBS,
    const void* __restrict__ Wp, int ldw,
    const float* __restrict__ bias, const float* __restrict__ bias2,
    const float* __restrict__ add, int addld, int addofs,
    int act, float cscale,
    float* __restrict__ C, unsigned short* __restrict__ C16,
    int cmode, int ldc, long cBS, int M, int N, int K)
{
  __shared__ unsigned short lA[TBM * AP];
  __shared__ unsigned short lB[TBK * BP];

  const int z = blockIdx.z, m0 = blockIdx.y * TBM, n0 = blockIdx.x * TBN;
  const int tid = threadIdx.x, wave = tid >> 5, lane = tid & 31;
  const int wm = (wave >> 2) * 32, wn = (wave & 3) * 32;
  const int mloc = lane & 15, half = lane >> 4;

  // hoisted A source pointer
  const int ar = tid >> 2, ac = (tid & 3) * 8;
  const int gm = m0 + ar;
  long arow;
  if (AMODE == 0)      arow = gm;
  else if (AMODE == 1) { int b = gm % Bb, l = gm / Bb; arow = (long)b * Lb + l; }
  else                 arow = (long)gm * Bb + z;
  const float*          aF = A16 ? nullptr
                                 : ((const float*)Ap + aBS * z + arow * lda + aofs + ac);
  const unsigned short* aH = A16 ? ((const unsigned short*)Ap + aBS * z + arow * lda + aofs + ac)
                                 : nullptr;
  const bool arowOK = (gm < M);

  // hoisted B source pointer for cooperative (BMODE==1) staging
  const int nIdx = tid >> 1, kc = (tid & 1) * 16;
  const unsigned short* bH = nullptr;
  if (BMODE == 1)
    bH = (const unsigned short*)Wp + ((long)(n0 + nIdx) * Bb + z) * ldw + kc;

  v8f acc[2][2] = {};

  for (int k0 = 0; k0 < K; k0 += TBK) {
    // ---- stage A ----
    if (arowOK) {
      if (A16) {
#pragma unroll
        for (int j = 0; j < 8; ++j) lA[ar * AP + ac + j] = aH[k0 + j];
      } else {
        if (k0 + TBK < K) __builtin_prefetch(aF + k0 + TBK, 0, 0);
#pragma unroll
        for (int j = 0; j < 8; ++j) lA[ar * AP + ac + j] = f2bf(aF[k0 + j]);
      }
    } else {
#pragma unroll
      for (int j = 0; j < 8; ++j) lA[ar * AP + ac + j] = 0;
    }
    // ---- stage B ----
    if (BTDM) {
      if (wave == 0) {
        const unsigned short* wb = (const unsigned short*)Wp;
        unsigned ldsoff = (unsigned)(uintptr_t)(&lB[0]);
        if (BMODE == 0)
          tdm_load_b_tile(ldsoff, wb, (long)k0 * ldw + n0, ldw,
                          (unsigned)ldw, (unsigned)K);
        else  // BMODE==2
          tdm_load_b_tile(ldsoff, wb, ((long)k0 * Bb + z) * ldw + n0,
                          (long)ldw * Bb, (unsigned)ldw, (unsigned)(K * Bb));
        __builtin_amdgcn_s_wait_tensorcnt(0);
      }
    } else {  // BMODE==1 cooperative copy (bf16 source)
#pragma unroll
      for (int j = 0; j < 16; ++j) lB[(kc + j) * BP + nIdx] = bH[k0 + j];
    }
    __syncthreads();

    // ---- fragments + WMMA ----
    Frag16 af[2], bf[2];
#pragma unroll
    for (int i = 0; i < 2; ++i) {
      int r = wm + i * 16 + mloc;
#pragma unroll
      for (int v = 0; v < 8; ++v) {
        int kk = ((v < 4) ? 0 : 16) + (half ? 8 : 0) + ((v & 3) << 1);
        af[i].u[v] = *(const unsigned*)&lA[r * AP + kk];
      }
    }
#pragma unroll
    for (int j = 0; j < 2; ++j)
#pragma unroll
      for (int v = 0; v < 16; ++v)
        bf[j].s[v] = lB[(((v < 8) ? 0 : 16) + (half ? 8 : 0) + (v & 7)) * BP + wn + j * 16 + mloc];
#pragma unroll
    for (int i = 0; i < 2; ++i)
#pragma unroll
      for (int j = 0; j < 2; ++j)
        acc[i][j] = __builtin_amdgcn_wmma_f32_16x16x32_bf16(
            false, af[i].v, false, bf[j].v, (short)0, acc[i][j], false, false);
    __syncthreads();
  }

  // ---- epilogue ----
#pragma unroll
  for (int i = 0; i < 2; ++i)
#pragma unroll
    for (int j = 0; j < 2; ++j) {
      v8f t = acc[i][j];
#pragma unroll
      for (int r = 0; r < 8; ++r) {
        int om = m0 + wm + i * 16 + r + half * 8;
        int on = n0 + wn + j * 16 + (lane & 15);
        if (om < M && on < N) {
          float v = t[r] * cscale;
          if (bias)  v += bias[on];
          if (bias2) v += bias2[on];
          long crow = (cmode == 0) ? (long)om : ((long)om * Bb + z);
          if (add)   v += add[crow * addld + addofs + on];
          if (act == 1) v = tanhf(v);
          else if (act == 2) v = fmaxf(v, 0.0f);
          if (C)   C[cBS * z + crow * ldc + on] = v;
          if (C16) C16[cBS * z + crow * ldc + on] = f2bf(v);
        }
      }
    }
}

// ======================= persistent LSTM scan ==============================
// 48 WGs x 16 h-cols; Whh slice (64 rows x 768 K, bf16, B-fragment swizzled)
// is LDS-resident for the whole 256-step scan.
__device__ __forceinline__ void grid_bar(unsigned* c, unsigned target) {
  __syncthreads();
  if (threadIdx.x == 0) {
    __threadfence();
    atomicAdd(c, 1u);
    while (atomicAdd(c, 0u) < target) __builtin_amdgcn_s_sleep(1);
  }
  __syncthreads();
}

__global__ __launch_bounds__(256) void k_lstm_scan(
    const float* __restrict__ Xg, const float* __restrict__ Whh,
    const float* __restrict__ imf1, float* __restrict__ hbuf,
    float* __restrict__ y, unsigned* __restrict__ counter, int T)
{
  const int NWG = 48;
  __shared__ unsigned short lW[24 * 32 * 68];  // [kchunk][klane][lrow(pad 68)]
  __shared__ unsigned short lH[16 * 776];
  __shared__ float lP[8 * 128];                // per-wave gate partials

  const int tid = threadIdx.x, j0 = blockIdx.x * 16;
  const int wave = tid >> 5, lane = tid & 31;
  const int mloc = lane & 15, half = lane >> 4;
  const int gate = wave >> 1, kh = wave & 1;
  const int urow = tid >> 4, ucol = tid & 15;  // update mapping (tid<128)

  // one-time: stage Whh slice (64 rows = 4 gates x 16 cols) bf16-swizzled
  for (int i = tid; i < 24 * 32 * 64; i += 256) {
    int lr = i & 63, t2 = i >> 6;
    int kk = t2 & 31, kc2 = t2 >> 5;
    int grow = (lr >> 4) * Hd + j0 + (lr & 15);
    lW[(kc2 * 32 + kk) * 68 + lr] = f2bf(Whh[(long)grow * Hd + kc2 * 32 + kk]);
  }
  for (int i = tid; i < 8 * 776; i += 256) lH[8 * 776 + i] = 0;  // zero rows 8..15

  float creg = 0.0f;
  if (tid < 128) {
    creg = imf1[urow * Hd + j0 + ucol];
    hbuf[urow * Hd + j0 + ucol] = creg;        // h0 = c0 = imf1
  }
  __threadfence();
  unsigned tgt = NWG;
  grid_bar(counter, tgt); tgt += NWG;

  for (int t = 0; t < T; ++t) {
    for (int i = tid; i < 8 * Hd; i += 256) {
      int r = i / Hd, c = i - r * Hd;
      lH[r * 776 + c] = f2bf(hbuf[i]);
    }
    __syncthreads();

    v8f acc = {};
    for (int kc = 0; kc < 12; ++kc) {
      int kc2 = kh * 12 + kc, k0 = kc2 * 32;
      Frag16 af, bfr;
#pragma unroll
      for (int v = 0; v < 8; ++v) {
        int kk = ((v < 4) ? 0 : 16) + (half ? 8 : 0) + ((v & 3) << 1);
        af.u[v] = *(const unsigned*)&lH[mloc * 776 + k0 + kk];
      }
#pragma unroll
      for (int v = 0; v < 16; ++v)
        bfr.s[v] = lW[(kc2 * 32 + ((v < 8) ? 0 : 16) + (half ? 8 : 0) + (v & 7)) * 68 + gate * 16 + mloc];
      acc = __builtin_amdgcn_wmma_f32_16x16x32_bf16(
          false, af.v, false, bfr.v, (short)0, acc, false, false);
    }
    if (!half) {
#pragma unroll
      for (int r = 0; r < 8; ++r) lP[wave * 128 + r * 16 + mloc] = acc[r];
    }
    __syncthreads();
    if (tid < 128) {
      int idx = urow * 16 + ucol;
      long xrow = ((long)t * Bb + urow) * 3072 + j0 + ucol;
      float gi = lP[0 * 128 + idx] + lP[1 * 128 + idx] + Xg[xrow + 0 * Hd];
      float gf = lP[2 * 128 + idx] + lP[3 * 128 + idx] + Xg[xrow + 1 * Hd];
      float gg = lP[4 * 128 + idx] + lP[5 * 128 + idx] + Xg[xrow + 2 * Hd];
      float go = lP[6 * 128 + idx] + lP[7 * 128 + idx] + Xg[xrow + 3 * Hd];
      creg = sigm(gf) * creg + sigm(gi) * tanhf(gg);
      float hv = sigm(go) * tanhf(creg);
      hbuf[urow * Hd + j0 + ucol] = hv;
      y[((long)t * Bb + urow) * Hd + j0 + ucol] = hv;
    }
    __threadfence();
    grid_bar(counter, tgt); tgt += NWG;
  }
}

// ======================= small helper kernels ==============================
// transposed bf16 weight: out[k*Nw + n] = bf16(W[(n+rowofs)*ldsrc + colofs + k]*s)
__global__ void k_cvt_t(const float* __restrict__ W, int ldsrc, int rowofs, int colofs,
                        int Nw, long total, float scale, unsigned short* __restrict__ out) {
  long g = blockIdx.x * 256L + threadIdx.x;
  if (g >= total) return;
  int n = (int)(g % Nw); long k = g / Nw;
  out[g] = f2bf(W[(long)(n + rowofs) * ldsrc + colofs + k] * scale);
}

__global__ void k_imgreduce(const float* __restrict__ img, float* __restrict__ imf1,
                            float* __restrict__ imf2m) {
  int g = blockIdx.x * 256 + threadIdx.x;
  if (g >= Bb * Hd) return;
  int b = g / Hd, c = g % Hd;
  float m1 = -1e30f, m2 = -1e30f;
  for (int n = 0; n < 197; ++n) {
    const float* row = img + (long)(b * 197 + n) * 1536;
    m1 = fmaxf(m1, row[c]); m2 = fmaxf(m2, row[Hd + c]);
  }
  imf1[g] = m1; imf2m[g] = m2;
}

__global__ void k_bcast(const float* __restrict__ s, float* __restrict__ d) {
  long g = blockIdx.x * 256L + threadIdx.x;
  if (g >= (long)Lb * Bb * Hd) return;
  long m = g / Hd; int c = g % Hd; int b = m % Bb;
  d[g] = s[b * Hd + c];
}

__global__ void k_colmaxL(const float* __restrict__ s, float* __restrict__ d) {
  int g = blockIdx.x * 256 + threadIdx.x;
  if (g >= Bb * Hd) return;
  int b = g / Hd, c = g % Hd;
  float m = -1e30f;
  for (int l = 0; l < Lb; ++l) m = fmaxf(m, s[((long)l * Bb + b) * Hd + c]);
  d[g] = m;
}

__global__ void k_concat2(const float* __restrict__ a, const float* __restrict__ b,
                          float* __restrict__ d) {
  long g = blockIdx.x * 256L + threadIdx.x;
  if (g >= (long)Lb * Bb * 1536) return;
  long m = g / 1536; int c = g % 1536;
  d[g] = (c < Hd) ? a[m * Hd + c] : b[m * Hd + c - Hd];
}

__global__ __launch_bounds__(256) void k_ln(
    const float* __restrict__ src, int srcld,
    const float* __restrict__ add, int addld, int addofs,
    const float* __restrict__ bias, const float* __restrict__ rscale,
    const float* __restrict__ g, const float* __restrict__ b,
    float* __restrict__ dst, int N)
{
  __shared__ float buf[768];
  __shared__ float red[256];
  int row = blockIdx.x, tid = threadIdx.x;
  float rs = rscale ? rscale[row % Bb] : 1.0f;
  float lsum = 0;
  for (int c = tid; c < N; c += 256) {
    float v = src[(long)row * srcld + c];
    if (bias) v += bias[c];
    v *= rs;
    if (add) v += add[(long)row * addld + addofs + c];
    buf[c] = v; lsum += v;
  }
  red[tid] = lsum; __syncthreads();
  for (int s = 128; s > 0; s >>= 1) { if (tid < s) red[tid] += red[tid + s]; __syncthreads(); }
  float mean = red[0] / N; __syncthreads();
  float lv = 0;
  for (int c = tid; c < N; c += 256) { float d = buf[c] - mean; lv += d * d; }
  red[tid] = lv; __syncthreads();
  for (int s = 128; s > 0; s >>= 1) { if (tid < s) red[tid] += red[tid + s]; __syncthreads(); }
  float rstd = rsqrtf(red[0] / N + 1e-5f); __syncthreads();
  for (int c = tid; c < N; c += 256)
    dst[(long)row * N + c] = (buf[c] - mean) * rstd * g[c] + b[c];
}

__global__ __launch_bounds__(256) void k_softmax(float* __restrict__ s) {
  __shared__ float red[256];
  int row = blockIdx.x, tid = threadIdx.x;
  float v = s[(long)row * 256 + tid];
  red[tid] = v; __syncthreads();
  for (int k = 128; k > 0; k >>= 1) { if (tid < k) red[tid] = fmaxf(red[tid], red[tid + k]); __syncthreads(); }
  float mx = red[0]; __syncthreads();
  float e = __expf(v - mx);
  red[tid] = e; __syncthreads();
  for (int k = 128; k > 0; k >>= 1) { if (tid < k) red[tid] += red[tid + k]; __syncthreads(); }
  s[(long)row * 256 + tid] = e / red[0];
}

__global__ void k_re_t(const float* __restrict__ hre1m, const float* __restrict__ hg,
                       const float* __restrict__ Wh, const float* __restrict__ bh,
                       float* __restrict__ t) {
  int g = blockIdx.x * 256 + threadIdx.x;
  if (g >= Bb * Hd) return;
  int b = g / Hd, j = g % Hd;
  const float* w = Wh + (long)j * 2304;
  float acc = bh[j];
  for (int k = 0; k < Hd; ++k) acc += w[k] * hre1m[b * Hd + k];
  for (int k = 0; k < Hd; ++k) acc += w[Hd + k] * hre1m[b * Hd + k];
  for (int k = 0; k < Hd; ++k) acc += w[2 * Hd + k] * hg[b * Hd + k];
  t[g] = acc;
}

__global__ __launch_bounds__(256) void k_re_lnelu(const float* __restrict__ t,
                                                  const float* __restrict__ g,
                                                  const float* __restrict__ bta,
                                                  float* __restrict__ u) {
  __shared__ float buf[768];
  __shared__ float red[256];
  int b = blockIdx.x, tid = threadIdx.x;
  float lsum = 0;
  for (int c = tid; c < Hd; c += 256) { float v = t[b * Hd + c]; buf[c] = v; lsum += v; }
  red[tid] = lsum; __syncthreads();
  for (int s = 128; s > 0; s >>= 1) { if (tid < s) red[tid] += red[tid + s]; __syncthreads(); }
  float mean = red[0] / Hd; __syncthreads();
  float lv = 0;
  for (int c = tid; c < Hd; c += 256) { float d = buf[c] - mean; lv += d * d; }
  red[tid] = lv; __syncthreads();
  for (int s = 128; s > 0; s >>= 1) { if (tid < s) red[tid] += red[tid + s]; __syncthreads(); }
  float rstd = rsqrtf(red[0] / Hd + 1e-5f); __syncthreads();
  for (int c = tid; c < Hd; c += 256) {
    float x = (buf[c] - mean) * rstd * g[c] + bta[c];
    u[b * Hd + c] = (x > 0.0f) ? x : (__expf(x) - 1.0f);
  }
}

__global__ void k_re_out(const float* __restrict__ u, const float* __restrict__ Wrel,
                         const float* __restrict__ brel, float* __restrict__ out) {
  int g = blockIdx.x * 256 + threadIdx.x;
  if (g >= Bb * 97) return;
  int b = g / 97, r = g % 97;
  float acc = brel[r];
  for (int j = 0; j < Hd; ++j) acc += u[b * Hd + j] * Wrel[(long)r * Hd + j];
  out[g] = acc;
}

// ======================= host orchestration ================================
template <int AM, int BMo, bool A16, bool B16, bool BT>
static void run_gemm(hipStream_t st, const void* A, int lda, int aofs, long aBS,
                     const void* W, int ldw,
                     const float* bias, const float* bias2,
                     const float* add, int addld, int addofs,
                     int act, float cscale,
                     float* C, unsigned short* C16, int cmode, int ldc, long cBS,
                     int M, int N, int K, int Z) {
  dim3 gr((N + TBN - 1) / TBN, (M + TBM - 1) / TBM, Z);
  k_gemm<AM, BMo, A16, B16, BT><<<gr, 256, 0, st>>>(
      A, lda, aofs, aBS, W, ldw, bias, bias2, add, addld, addofs,
      act, cscale, C, C16, cmode, ldc, cBS, M, N, K);
}

extern "C" void kernel_launch(void* const* d_in, const int* in_sizes, int n_in,
                              void* d_out, int out_size, void* d_ws, size_t ws_size,
                              hipStream_t stream) {
  (void)in_sizes; (void)n_in; (void)out_size; (void)ws_size;
  const float DC = 0.9f;
  const float SCAL = 0.03608439182435161f;  // 1/sqrt(768)

  const float* x     = (const float*)d_in[0];
  const float* imgf  = (const float*)d_in[1];
  const float* sim   = (const float*)d_in[2];
  const float* W_img = (const float*)d_in[4];
  const float* b_img = (const float*)d_in[5];
  const float* W_txt = (const float*)d_in[6];
  const float* b_txt = (const float*)d_in[7];
  const float* Wih0  = (const float*)d_in[8];
  const float* Whh0  = (const float*)d_in[9];
  const float* bih0  = (const float*)d_in[10];
  const float* bhh0  = (const float*)d_in[11];
  const float* Wih1  = (const float*)d_in[12];
  const float* Whh1  = (const float*)d_in[13];
  const float* bih1  = (const float*)d_in[14];
  const float* bhh1  = (const float*)d_in[15];
  const float* Wr    = (const float*)d_in[48];
  const float* br    = (const float*)d_in[49];
  const float* Wh    = (const float*)d_in[50];
  const float* bh    = (const float*)d_in[51];
  const float* gre   = (const float*)d_in[52];
  const float* bre   = (const float*)d_in[53];
  const float* Wrel  = (const float*)d_in[54];
  const float* brel  = (const float*)d_in[55];

  unsigned* cnt0 = (unsigned*)d_ws;
  unsigned* cnt1 = cnt0 + 16;
  float* wsf = (float*)d_ws;
  long off = 64;
  auto alloc = [&](long n) { float* p = wsf + off; off += (n + 63) & ~63L; return p; };
  auto alloc16 = [&](long n) { return (unsigned short*)alloc((n + 1) / 2); };

  float* img     = alloc(1576L * 1536);
  float* imf1    = alloc(Bb * Hd);
  float* imf2m   = alloc(Bb * Hd);
  float* hbig    = alloc(2048L * 1536);
  float* Xg      = alloc(2048L * 3072);
  float* ybuf    = alloc(2048L * Hd);
  float* h_re1   = alloc(2048L * Hd);
  float* hstate  = alloc(Bb * Hd);
  float* scores  = alloc(8L * 256 * 256);
  float* Obuf    = alloc(2048L * Hd);
  float* x1      = alloc(2048L * Hd);
  float* ffb     = alloc(2048L * Hd);
  float* selfout = alloc(2048L * Hd);
  float* qcross  = alloc(2048L * Hd);
  float* h_re2   = alloc(2048L * Hd);
  float* cat     = alloc(2048L * 1536);
  float* hgfull  = alloc(2048L * Hd);
  float* hg      = alloc(Bb * Hd);
  float* hre1m   = alloc(Bb * Hd);
  float* tbuf    = alloc(Bb * Hd);
  float* ubuf    = alloc(Bb * Hd);
  unsigned short* WimgT = alloc16(768L * 1536);
  unsigned short* WtxtT = alloc16(768L * 1536);
  unsigned short* Wih0T = alloc16(768L * 3072);
  unsigned short* Wih1T = alloc16(768L * 3072);
  unsigned short* Wf1T  = alloc16(768L * 768);
  unsigned short* Wf2T  = alloc16(768L * 768);
  unsigned short* WrT   = alloc16(1536L * 768);
  unsigned short* WqTh  = alloc16(768L * 768);
  unsigned short* WkTh  = alloc16(768L * 768);
  unsigned short* WvTh  = alloc16(768L * 768);
  unsigned short* WoTh  = alloc16(768L * 768);
  unsigned short* Qh16  = alloc16(2048L * Hd);
  unsigned short* Kh16  = alloc16(2048L * Hd);
  unsigned short* Vh16  = alloc16(2048L * Hd);
  unsigned short* AOh16 = alloc16(2048L * Hd);

  (void)hipMemsetAsync(d_ws, 0, 256, stream);

  auto cvt = [&](const float* W, int ldsrc, int rowofs, int colofs, int Nw, long Kw,
                 float scale, unsigned short* out) {
    long total = (long)Nw * Kw;
    k_cvt_t<<<dim3((unsigned)((total + 255) / 256)), 256, 0, stream>>>(
        W, ldsrc, rowofs, colofs, Nw, total, scale, out);
  };

  // weight conversions (transposed bf16 [K,N]); DropConnect folded in
  cvt(W_img, 768, 0, 0, 1536, 768, DC, WimgT);
  cvt(W_txt, 768, 0, 0, 1536, 768, DC, WtxtT);
  cvt(Wih0, 768, 0, 0, 3072, 768, 1.0f, Wih0T);
  cvt(Wih1, 768, 0, 0, 3072, 768, 1.0f, Wih1T);
  cvt(Wr, 1536, 0, 0, 768, 1536, 1.0f, WrT);

  // 1) img = tanh(img_feat @ (W_img*DC)^T + b_img)  -> max-reduce
  run_gemm<0, 0, false, true, true>(stream, imgf, 768, 0, 0, WimgT, 1536,
      b_img, nullptr, nullptr, 0, 0, 1, 1.0f, img, nullptr, 0, 1536, 0,
      1576, 1536, 768, 1);
  k_imgreduce<<<(Bb * Hd + 255) / 256, 256, 0, stream>>>(img, imf1, imf2m);

  // 2) h = x^T @ (W_txt*DC)^T + b_txt   (rows m = l*B + b)
  run_gemm<1, 0, false, true, true>(stream, x, 768, 0, 0, WtxtT, 1536,
      b_txt, nullptr, nullptr, 0, 0, 0, 1.0f, hbig, nullptr, 0, 1536, 0,
      2048, 1536, 768, 1);

  // 3) LSTM layers
  run_gemm<0, 0, false, true, true>(stream, hbig, 1536, 0, 0, Wih0T, 3072,
      bih0, bhh0, nullptr, 0, 0, 0, 1.0f, Xg, nullptr, 0, 3072, 0,
      2048, 3072, 768, 1);
  k_lstm_scan<<<48, 256, 0, stream>>>(Xg, Whh0, imf1, hstate, ybuf, cnt0, Lb);
  run_gemm<0, 0, false, true, true>(stream, ybuf, 768, 0, 0, Wih1T, 3072,
      bih1, bhh1, nullptr, 0, 0, 0, 1.0f, Xg, nullptr, 0, 3072, 0,
      2048, 3072, 768, 1);
  k_lstm_scan<<<48, 256, 0, stream>>>(Xg, Whh1, imf1, hstate, h_re1, cnt1, Lb);

  auto layer = [&](const float* qin, int qld, int qofs,
                   const float* kv, int kvld, int kvofs,
                   int pb, const float* rscale, float* outbuf) {
    const float* Wq  = (const float*)d_in[pb + 0];
    const float* bq  = (const float*)d_in[pb + 1];
    const float* Wk  = (const float*)d_in[pb + 2];
    const float* bk  = (const float*)d_in[pb + 3];
    const float* Wv  = (const float*)d_in[pb + 4];
    const float* bv  = (const float*)d_in[pb + 5];
    const float* Wo  = (const float*)d_in[pb + 6];
    const float* bo  = (const float*)d_in[pb + 7];
    const float* Wf1 = (const float*)d_in[pb + 8];
    const float* bf1 = (const float*)d_in[pb + 9];
    const float* Wf2 = (const float*)d_in[pb + 10];
    const float* bf2 = (const float*)d_in[pb + 11];
    const float* g1  = (const float*)d_in[pb + 12];
    const float* b1  = (const float*)d_in[pb + 13];
    const float* g2  = (const float*)d_in[pb + 14];
    const float* b2  = (const float*)d_in[pb + 15];

    for (int hd = 0; hd < 12; ++hd) {
      cvt(Wq, 768, hd * Hd, 0, 768, 768, 1.0f, WqTh);
      cvt(Wk, 768, hd * Hd, 0, 768, 768, 1.0f, WkTh);
      cvt(Wv, 768, hd * Hd, 0, 768, 768, 1.0f, WvTh);
      cvt(Wo, 9216, 0, hd * Hd, 768, 768, 1.0f, WoTh);
      // Qh/Kh/Vh written directly as bf16
      run_gemm<0, 0, false, true, true>(stream, qin, qld, qofs, 0, WqTh, 768,
          bq + hd * Hd, nullptr, nullptr, 0, 0, 0, 1.0f,
          nullptr, Qh16, 0, 768, 0, 2048, 768, 768, 1);
      run_gemm<0, 0, false, true, true>(stream, kv, kvld, kvofs, 0, WkTh, 768,
          bk + hd * Hd, nullptr, nullptr, 0, 0, 0, 1.0f,
          nullptr, Kh16, 0, 768, 0, 2048, 768, 768, 1);
      run_gemm<0, 0, false, true, true>(stream, kv, kvld, kvofs, 0, WvTh, 768,
          bv + hd * Hd, nullptr, nullptr, 0, 0, 0, 1.0f,
          nullptr, Vh16, 0, 768, 0, 2048, 768, 768, 1);
      // scores[z] = (Qh @ Kh^T) / sqrt(768)
      run_gemm<2, 1, true, true, false>(stream, Qh16, 768, 0, 0, Kh16, 768,
          nullptr, nullptr, nullptr, 0, 0, 0, SCAL,
          scores, nullptr, 0, 256, 65536L, 256, 256, 768, 8);
      k_softmax<<<2048, 256, 0, stream>>>(scores);
      // AOh = softmax(scores) @ Vh  (rows back to m = l*B + b)
      run_gemm<0, 2, false, true, true>(stream, scores, 256, 0, 65536L, Vh16, 768,
          nullptr, nullptr, nullptr, 0, 0, 0, 1.0f,
          nullptr, AOh16, 1, 768, 0, 256, 768, 256, 8);
      // O (+)= AOh @ Wo_head^T
      run_gemm<0, 0, true, true, true>(stream, AOh16, 768, 0, 0, WoTh, 768,
          nullptr, nullptr, (hd ? Obuf : nullptr), 768, 0, 0, 1.0f,
          Obuf, nullptr, 0, 768, 0, 2048, 768, 768, 1);
    }
    k_ln<<<2048, 256, 0, stream>>>(Obuf, 768, kv, kvld, kvofs, bo, rscale, g1, b1, x1, 768);
    cvt(Wf1, 768, 0, 0, 768, 768, 1.0f, Wf1T);
    cvt(Wf2, 768, 0, 0, 768, 768, 1.0f, Wf2T);
    run_gemm<0, 0, false, true, true>(stream, x1, 768, 0, 0, Wf1T, 768,
        bf1, nullptr, nullptr, 0, 0, 2, 1.0f, ffb, nullptr, 0, 768, 0,
        2048, 768, 768, 1);
    run_gemm<0, 0, false, true, true>(stream, ffb, 768, 0, 0, Wf2T, 768,
        bf2, nullptr, x1, 768, 0, 0, 1.0f, Obuf, nullptr, 0, 768, 0,
        2048, 768, 768, 1);
    k_ln<<<2048, 256, 0, stream>>>(Obuf, 768, nullptr, 0, 0, nullptr, nullptr, g2, b2,
                                   outbuf, 768);
  };

  // 5) self layer: q = h_re1, kv = h2 (cols 768..1535 of hbig)
  layer(h_re1, 768, 0, hbig, 1536, 768, 16, nullptr, selfout);

  // 6) cross layer: q = broadcast(imf2max), kv = selfout, scaled by sim
  k_bcast<<<(2048 * 768 + 255) / 256, 256, 0, stream>>>(imf2m, qcross);
  layer(qcross, 768, 0, selfout, 768, 0, 32, sim, h_re2);

  // 7) RE head
  k_concat2<<<(2048 * 1536 + 255) / 256, 256, 0, stream>>>(h_re1, h_re2, cat);
  run_gemm<0, 0, false, true, true>(stream, cat, 1536, 0, 0, WrT, 768,
      br, nullptr, nullptr, 0, 0, 1, 1.0f, hgfull, nullptr, 0, 768, 0,
      2048, 768, 1536, 1);
  k_colmaxL<<<(Bb * Hd + 255) / 256, 256, 0, stream>>>(hgfull, hg);
  k_colmaxL<<<(Bb * Hd + 255) / 256, 256, 0, stream>>>(h_re1, hre1m);
  k_re_t<<<(Bb * Hd + 255) / 256, 256, 0, stream>>>(hre1m, hg, Wh, bh, tbuf);
  k_re_lnelu<<<Bb, 256, 0, stream>>>(tbuf, gre, bre, ubuf);
  k_re_out<<<(Bb * 97 + 255) / 256, 256, 0, stream>>>(ubuf, Wrel, brel, (float*)d_out);
}
